// PagedAttention_78262894068470
// MI455X (gfx1250) — compile-verified, matrix operand order checked by reading, not run on hardware
//
#include <hip/hip_runtime.h>
#include <hip/hip_bf16.h>

// ---------------------------------------------------------------------------
// Paged attention for MI455X (gfx1250, wave32, WMMA).
// B=4, S=4096, HID=2048, H=16, P=256, D=128, N=16.
// ---------------------------------------------------------------------------

constexpr int kB   = 4;
constexpr int kS   = 4096;
constexpr int kHid = 2048;
constexpr int kH   = 16;
constexpr int kP   = 256;
constexpr int kD   = 128;
constexpr int kN   = 16;
constexpr int kM   = kB * kS;          // 16384 rows in the projection GEMMs

typedef __attribute__((ext_vector_type(16))) __bf16 v16bf;
typedef __attribute__((ext_vector_type(8)))  float  v8f;

union FragU { unsigned int u[8]; v16bf v; };

__device__ __forceinline__ unsigned short f2bf(float x) {
    union { float f; unsigned int u; } c; c.f = x;
    unsigned int r = c.u + 0x7FFFu + ((c.u >> 16) & 1u);   // round-to-nearest-even
    return (unsigned short)(r >> 16);
}

__device__ __forceinline__ v8f v8f_zero() {
    v8f z;
#pragma unroll
    for (int i = 0; i < 8; ++i) z[i] = 0.0f;
    return z;
}

__device__ __forceinline__ v8f wmma_bf16(v16bf a, v16bf b, v8f c) {
    // D = A(16x32 bf16) * B(32x16 bf16) + C(16x16 f32)
    return __builtin_amdgcn_wmma_f32_16x16x32_bf16(
        /*neg_a=*/false, a, /*neg_b=*/false, b,
        /*c_mod=*/(short)0, c, /*reuse_a=*/false, /*reuse_b=*/false);
}

// A-fragment (16x32 bf16) from row-major LDS; also B-fragment when B == X^T of
// a row-major X (lane-mirrored layout). ISA 7.12.2: lanes 0-15 row=lane,
// K pairs {0..7,16..23}; lanes 16-31 row=lane-16, K pairs {8..15,24..31}.
__device__ __forceinline__ v16bf load_frag_row(const unsigned short* base, int stride, int lane) {
    const unsigned short* p = base + (lane & 15) * stride + ((lane & 16) ? 8 : 0);
    FragU f;
#pragma unroll
    for (int j = 0; j < 4; ++j) {
        f.u[j]     = *(const unsigned int*)(p + 2 * j);
        f.u[j + 4] = *(const unsigned int*)(p + 16 + 2 * j);
    }
    return f.v;
}

// B-fragment (32x16 bf16) from a row-major K x N LDS tile (stride = row pitch),
// base pre-offset to (k0, n0). Lane holds column n = lane%16.
__device__ __forceinline__ v16bf load_frag_col(const unsigned short* base, int stride, int lane) {
    const unsigned short* p = base + (lane & 15) + ((lane & 16) ? 8 : 0) * stride;
    FragU f;
#pragma unroll
    for (int j = 0; j < 4; ++j) {
        f.u[j]     = (unsigned int)p[(2 * j) * stride]      | ((unsigned int)p[(2 * j + 1) * stride] << 16);
        f.u[j + 4] = (unsigned int)p[(16 + 2 * j) * stride] | ((unsigned int)p[(17 + 2 * j) * stride] << 16);
    }
    return f.v;
}

// ---------------------------------------------------------------------------
// QKV projection: Y[s,c] = sum_k X[s,k] * W[c,k] + bias[c]
// X fp32 (16384 x 2048) row-major, W fp32 (2048 x 2048) row-major.
// Output written TRANSPOSED as bf16: OutT[b][c][s_local]  (B, HID, S)
// so attention Q'/K'/V' rows become contiguous.
// ---------------------------------------------------------------------------
__global__ __launch_bounds__(256) void gemm_qkv_kernel(const float* __restrict__ X,
                                                       const float* __restrict__ W,
                                                       const float* __restrict__ bias,
                                                       unsigned short* __restrict__ OutT) {
    __shared__ unsigned short As[128 * 32];
    __shared__ unsigned short Bs[128 * 32];

    const int m0 = blockIdx.x * 128;
    const int n0 = blockIdx.y * 128;
    const int t = threadIdx.x;
    const int lane = t & 31;
    const int wave = t >> 5;
    const int wm = wave >> 1;   // 0..3 -> 32-row strip
    const int wn = wave & 1;    // 0..1 -> 64-col strip

    v8f acc[2][4];
#pragma unroll
    for (int i = 0; i < 2; ++i)
#pragma unroll
        for (int j = 0; j < 4; ++j) acc[i][j] = v8f_zero();

    const int r = t >> 1, c0 = (t & 1) * 16;   // each thread fills 16 elems/row-half
    for (int kb = 0; kb < kHid; kb += 32) {
        __syncthreads();
        const float* sa = X + (size_t)(m0 + r) * kHid + kb + c0;
        const float* sb = W + (size_t)(n0 + r) * kHid + kb + c0;
        unsigned short* da = As + r * 32 + c0;
        unsigned short* db = Bs + r * 32 + c0;
#pragma unroll
        for (int i = 0; i < 16; ++i) { da[i] = f2bf(sa[i]); db[i] = f2bf(sb[i]); }
        __syncthreads();

        v16bf af[2], bfrag[4];
#pragma unroll
        for (int i = 0; i < 2; ++i) af[i] = load_frag_row(As + (wm * 32 + i * 16) * 32, 32, lane);
#pragma unroll
        for (int j = 0; j < 4; ++j) bfrag[j] = load_frag_row(Bs + (wn * 64 + j * 16) * 32, 32, lane);
#pragma unroll
        for (int i = 0; i < 2; ++i)
#pragma unroll
            for (int j = 0; j < 4; ++j) acc[i][j] = wmma_bf16(af[i], bfrag[j], acc[i][j]);
    }

    // Transposed bf16 epilogue. All 128 rows of this block share one batch idx.
    const int half = lane >> 4, nl = lane & 15;
    const int bidx = m0 >> 12;                     // m0 / 4096
#pragma unroll
    for (int i = 0; i < 2; ++i) {
        const int sgb = m0 + wm * 32 + i * 16 + half * 8;
        const int slb = sgb & (kS - 1);
#pragma unroll
        for (int j = 0; j < 4; ++j) {
            const int cg = n0 + wn * 64 + j * 16 + nl;
            const float bv = bias[cg];
            unsigned short* dst = OutT + ((size_t)(bidx * kHid + cg)) * kS + slb;
#pragma unroll
            for (int jj = 0; jj < 8; ++jj) dst[jj] = f2bf(acc[i][j][jj] + bv);
        }
    }
}

// ---------------------------------------------------------------------------
// Attention over one paged block (b, h, n).
// Q'[p2][d2] = QT[b][h*128 + p2/2][n*256 + (p2&1)*128 + d2]   (same for K', V')
// scores = Q' K'^T * SCALE ; softmax rows ; ctx = P V'
// ctx written bf16 row-major as (B, 4096, 2048): row = h*256+p2, col = n*128+d2.
// LDS: Q'(64K) + K'(64K) + V'(64K) + per-wave probs (64K) = 256 KB.
// ---------------------------------------------------------------------------
__global__ __launch_bounds__(256) void attn_kernel(const unsigned short* __restrict__ QT,
                                                   const unsigned short* __restrict__ KT,
                                                   const unsigned short* __restrict__ VT,
                                                   unsigned short* __restrict__ Ctx) {
    extern __shared__ uint4 smv[];
    unsigned short* Qs = (unsigned short*)smv;     // 256 x 128
    unsigned short* Ks = Qs + 256 * 128;
    unsigned short* Vs = Ks + 256 * 128;
    unsigned short* Ps = Vs + 256 * 128;           // 8 waves x (16 x 256)

    const int blk = blockIdx.x;                    // b*256 + h*16 + n
    const int n = blk & 15;
    const int h = (blk >> 4) & 15;
    const int b = blk >> 8;
    const int t = threadIdx.x, lane = t & 31, wave = t >> 5;

    // Cooperative staging: 3 x 256 rows x 256B, 16B chunks, coalesced.
    const size_t cb = (size_t)b * kHid * kS;
#pragma unroll
    for (int j = 0; j < 16; ++j) {
        const int idx = t + 256 * j;               // 4096 chunks per matrix
        const int row = idx >> 4, c16 = idx & 15;
        const size_t g = cb + (size_t)(h * kD + (row >> 1)) * kS
                            + n * kP + (row & 1) * 128 + c16 * 8;
        const int l = row * 128 + c16 * 8;
        *(uint4*)(Qs + l) = *(const uint4*)(QT + g);
        *(uint4*)(Ks + l) = *(const uint4*)(KT + g);
        *(uint4*)(Vs + l) = *(const uint4*)(VT + g);
    }
    __syncthreads();

    const float SCALE = 0.08838834764831845f;      // 128^-0.5
    const int half = lane >> 4, nl = lane & 15;
    unsigned short* Pw = Ps + wave * 16 * 256;

    for (int pass = 0; pass < 2; ++pass) {
        const int mt = wave + pass * 8;            // 16-row strip index (0..15)
        const unsigned short* Arow = Qs + mt * 16 * 128;

        // ---- scores: 16 column-tiles, K = 128 ----
        v8f sc[16];
#pragma unroll
        for (int nt = 0; nt < 16; ++nt) sc[nt] = v8f_zero();
        for (int k0 = 0; k0 < 128; k0 += 32) {
            v16bf a = load_frag_row(Arow + k0, 128, lane);
#pragma unroll
            for (int nt = 0; nt < 16; ++nt) {
                v16bf kf = load_frag_row(Ks + nt * 16 * 128 + k0, 128, lane);
                sc[nt] = wmma_bf16(a, kf, sc[nt]);
            }
        }

        // ---- row softmax (C-layout: lanes 0-15 rows j, lanes 16-31 rows j+8) ----
#pragma unroll
        for (int j = 0; j < 8; ++j) {
            float m = -3.0e38f;
#pragma unroll
            for (int nt = 0; nt < 16; ++nt) m = fmaxf(m, sc[nt][j]);
            for (int s = 1; s < 16; s <<= 1) m = fmaxf(m, __shfl_xor(m, s, 32));
            float sum = 0.0f;
#pragma unroll
            for (int nt = 0; nt < 16; ++nt) {
                float p = __expf((sc[nt][j] - m) * SCALE);
                sc[nt][j] = p;
                sum += p;
            }
            for (int s = 1; s < 16; s <<= 1) sum += __shfl_xor(sum, s, 32);
            const float inv = 1.0f / sum;
            const int rr = j + 8 * half;
#pragma unroll
            for (int nt = 0; nt < 16; ++nt)
                Pw[rr * 256 + nt * 16 + nl] = f2bf(sc[nt][j] * inv);   // lane transpose via LDS
        }

        // ---- ctx = P (16x256) x V' (256x128) ----
        v8f cacc[8];
#pragma unroll
        for (int j = 0; j < 8; ++j) cacc[j] = v8f_zero();
        for (int k0 = 0; k0 < 256; k0 += 32) {
            v16bf a = load_frag_row(Pw + k0, 256, lane);
#pragma unroll
            for (int nt2 = 0; nt2 < 8; ++nt2) {
                v16bf vf = load_frag_col(Vs + k0 * 128 + nt2 * 16, 128, lane);
                cacc[nt2] = wmma_bf16(a, vf, cacc[nt2]);
            }
        }

        // ---- write ctx (bf16, row-major for the output GEMM) ----
#pragma unroll
        for (int nt2 = 0; nt2 < 8; ++nt2) {
            const int d2 = nt2 * 16 + nl;
            const size_t col = (size_t)(n * kD + d2);
#pragma unroll
            for (int j = 0; j < 8; ++j) {
                const int p2q = mt * 16 + j + 8 * half;
                const size_t rowg = (size_t)b * kS + h * kP + p2q;
                Ctx[rowg * kHid + col] = f2bf(cacc[nt2][j]);
            }
        }
    }
}

// ---------------------------------------------------------------------------
// Output projection: Out[s,c] = sum_k Ctx[s,k] * Wo[c,k] + bo[c], fp32 out.
// ---------------------------------------------------------------------------
__global__ __launch_bounds__(256) void gemm_out_kernel(const unsigned short* __restrict__ Actx,
                                                       const float* __restrict__ W,
                                                       const float* __restrict__ bias,
                                                       float* __restrict__ Out) {
    __shared__ unsigned short As[128 * 32];
    __shared__ unsigned short Bs[128 * 32];

    const int m0 = blockIdx.x * 128;
    const int n0 = blockIdx.y * 128;
    const int t = threadIdx.x;
    const int lane = t & 31;
    const int wave = t >> 5;
    const int wm = wave >> 1;
    const int wn = wave & 1;

    v8f acc[2][4];
#pragma unroll
    for (int i = 0; i < 2; ++i)
#pragma unroll
        for (int j = 0; j < 4; ++j) acc[i][j] = v8f_zero();

    const int r = t >> 1, c0 = (t & 1) * 16;
    for (int kb = 0; kb < kHid; kb += 32) {
        __syncthreads();
        const unsigned short* sa = Actx + (size_t)(m0 + r) * kHid + kb + c0;
        const float* sb = W + (size_t)(n0 + r) * kHid + kb + c0;
        unsigned short* da = As + r * 32 + c0;
        unsigned short* db = Bs + r * 32 + c0;
#pragma unroll
        for (int i = 0; i < 16; ++i) { da[i] = sa[i]; db[i] = f2bf(sb[i]); }
        __syncthreads();

        v16bf af[2], bfrag[4];
#pragma unroll
        for (int i = 0; i < 2; ++i) af[i] = load_frag_row(As + (wm * 32 + i * 16) * 32, 32, lane);
#pragma unroll
        for (int j = 0; j < 4; ++j) bfrag[j] = load_frag_row(Bs + (wn * 64 + j * 16) * 32, 32, lane);
#pragma unroll
        for (int i = 0; i < 2; ++i)
#pragma unroll
            for (int j = 0; j < 4; ++j) acc[i][j] = wmma_bf16(af[i], bfrag[j], acc[i][j]);
    }

    const int half = lane >> 4, nl = lane & 15;
#pragma unroll
    for (int i = 0; i < 2; ++i) {
        const int sgb = m0 + wm * 32 + i * 16 + half * 8;
#pragma unroll
        for (int j = 0; j < 4; ++j) {
            const int cg = n0 + wn * 64 + j * 16 + nl;
            const float bv = bias[cg];
#pragma unroll
            for (int jj = 0; jj < 8; ++jj)
                Out[(size_t)(sgb + jj) * kHid + cg] = acc[i][j][jj] + bv;
        }
    }
}

// ---------------------------------------------------------------------------
extern "C" void kernel_launch(void* const* d_in, const int* in_sizes, int n_in,
                              void* d_out, int out_size, void* d_ws, size_t ws_size,
                              hipStream_t stream) {
    const float* hidden = (const float*)d_in[0];
    const float* Wq = (const float*)d_in[1];
    const float* bq = (const float*)d_in[2];
    const float* Wk = (const float*)d_in[3];
    const float* bk = (const float*)d_in[4];
    const float* Wv = (const float*)d_in[5];
    const float* bv = (const float*)d_in[6];
    const float* Wo = (const float*)d_in[7];
    const float* bo = (const float*)d_in[8];
    float* out = (float*)d_out;

    char* ws = (char*)d_ws;
    const size_t szT = (size_t)kB * kHid * kS * sizeof(unsigned short);   // 64 MiB each
    unsigned short* QT  = (unsigned short*)(ws + 0 * szT);
    unsigned short* KT  = (unsigned short*)(ws + 1 * szT);
    unsigned short* VT  = (unsigned short*)(ws + 2 * szT);
    unsigned short* Ctx = (unsigned short*)(ws + 3 * szT);

    dim3 gg(kM / 128, kHid / 128);   // 128 x 16 tiles
    gemm_qkv_kernel<<<gg, 256, 0, stream>>>(hidden, Wq, bq, QT);
    gemm_qkv_kernel<<<gg, 256, 0, stream>>>(hidden, Wk, bk, KT);
    gemm_qkv_kernel<<<gg, 256, 0, stream>>>(hidden, Wv, bv, VT);

    attn_kernel<<<kB * kH * kN, 256, 256 * 1024, stream>>>(QT, KT, VT, Ctx);

    gemm_out_kernel<<<gg, 256, 0, stream>>>(Ctx, Wo, bo, out);
}
